// QuerySelector_19868518711631
// MI455X (gfx1250) — matrix operand order repaired and checked by reading, not compiled, not run on hardware
//
#include <hip/hip_runtime.h>

// ---------------- problem constants ----------------
#define B_  8
#define LQ_ 1024
#define S_  256
#define D_  1024
#define H_  16
#define HD_ 64
#define LN_EPS 1e-5f

typedef __attribute__((ext_vector_type(16))) __bf16 v16bf;
typedef __attribute__((ext_vector_type(8)))  float  v8f;

union Frag { unsigned int u[8]; v16bf v; };

__device__ __forceinline__ unsigned short f2bf(float f) {
    union { float f; unsigned int u; } x; x.f = f;
    unsigned int u = x.u;
    unsigned int r = (u + 0x7FFFu + ((u >> 16) & 1u)) >> 16;  // RNE
    return (unsigned short)r;
}
__device__ __forceinline__ float bf2f(unsigned short s) {
    union { unsigned int u; float f; } x; x.u = ((unsigned int)s) << 16;
    return x.f;
}

// A-fragment K-base for vgpr j, lane-half hi (contiguous k = 16*hi + 2*j, matches B loads)
__device__ __forceinline__ int a_kbase(int j, int hi) {
    return hi * 16 + 2 * j;
}

#define WMMA_BF16(A, Bv, C) \
    __builtin_amdgcn_wmma_f32_16x16x32_bf16(false, (A), false, (Bv), (short)0, (C), false, false)

// ---------------- prep kernels ----------------
__global__ void pack_combined_kernel(const float* __restrict__ mem,
                                     const float* __restrict__ gaze,
                                     unsigned short* __restrict__ out) {
    int idx = blockIdx.x * blockDim.x + threadIdx.x;   // over B*LQ*D
    int col = idx & (D_ - 1);
    int row = idx >> 10;
    float v = (col < D_ / 2) ? mem[row * (D_ / 2) + col]
                             : gaze[row * (D_ / 2) + (col - D_ / 2)];
    out[idx] = f2bf(v);
}

__global__ void cvt_f32_bf16_kernel(const float* __restrict__ in,
                                    unsigned short* __restrict__ out, int n) {
    int idx = blockIdx.x * blockDim.x + threadIdx.x;
    if (idx < n) out[idx] = f2bf(in[idx]);
}

// ---------------- generic  Out = A(Mx1024,bf16) @ W(1024x1024,bf16)^T + bias ----------------
// block: 256 thr (8 waves); wave tile 16x64 (1 A-frag x 4 B-frags, 4 wmma/k-step);
// block tile 128x64. Double-buffered fragment prefetch (fits ~112 VGPRs, no spills).
// vt!=0: scatter-store output as V^T per head: VT[((b*H+h)*HD+hd)*S + s].
__global__ void gemm_xwT_kernel(const unsigned short* __restrict__ A,
                                const unsigned short* __restrict__ W,
                                const float* __restrict__ bias,
                                unsigned short* __restrict__ OutBf,
                                float* __restrict__ OutF,
                                int relu, int vt) {
    const int K = D_;      // compile-time K == N == 1024
    int tid  = threadIdx.x;
    int lane = tid & 31, wv = tid >> 5;
    int hi   = lane >> 4, ln = lane & 15;
    int row0 = blockIdx.x * 128 + wv * 16;
    int col0 = blockIdx.y * 64;

    const unsigned short* arow = A + (size_t)(row0 + ln) * K;
    const unsigned short* wr0 = W + (size_t)(col0 +  0 + ln) * K + hi * 16;
    const unsigned short* wr1 = W + (size_t)(col0 + 16 + ln) * K + hi * 16;
    const unsigned short* wr2 = W + (size_t)(col0 + 32 + ln) * K + hi * 16;
    const unsigned short* wr3 = W + (size_t)(col0 + 48 + ln) * K + hi * 16;

    auto loadA = [&](Frag& d, int k0) {
#pragma unroll
        for (int j = 0; j < 8; ++j)
            d.u[j] = *(const unsigned int*)(arow + k0 + a_kbase(j, hi));
    };
    auto loadB = [&](Frag* d, int k0) {
#pragma unroll
        for (int j = 0; j < 8; ++j) {
            d[0].u[j] = *(const unsigned int*)(wr0 + k0 + 2 * j);
            d[1].u[j] = *(const unsigned int*)(wr1 + k0 + 2 * j);
            d[2].u[j] = *(const unsigned int*)(wr2 + k0 + 2 * j);
            d[3].u[j] = *(const unsigned int*)(wr3 + k0 + 2 * j);
        }
    };

    v8f acc[4] = {};
    Frag a0, a1, b0[4], b1[4];
    loadA(a0, 0);
    loadB(b0, 0);
    for (int k0 = 0; k0 < K; k0 += 64) {            // K multiple of 64
        loadA(a1, k0 + 32);
        loadB(b1, k0 + 32);
#pragma unroll
        for (int nf = 0; nf < 4; ++nf)
            acc[nf] = WMMA_BF16(a0.v, b0[nf].v, acc[nf]);
        if (k0 + 64 < K) {
            loadA(a0, k0 + 64);
            loadB(b0, k0 + 64);
        }
#pragma unroll
        for (int nf = 0; nf < 4; ++nf)
            acc[nf] = WMMA_BF16(a1.v, b1[nf].v, acc[nf]);
    }

#pragma unroll
    for (int nf = 0; nf < 4; ++nf)
#pragma unroll
        for (int j = 0; j < 8; ++j) {
            int row = row0 + j + hi * 8;
            int col = col0 + nf * 16 + ln;
            float v = acc[nf][j] + bias[col];
            if (relu) v = fmaxf(v, 0.0f);
            if (OutF) {
                OutF[(size_t)row * D_ + col] = v;
            } else if (vt) {
                int bb = row >> 8, s = row & (S_ - 1);   // row = b*S + s
                int hh = col >> 6, hd = col & (HD_ - 1); // col = h*HD + hd
                OutBf[((size_t)(bb * H_ + hh) * HD_ + hd) * S_ + s] = f2bf(v);
            } else {
                OutBf[(size_t)row * D_ + col] = f2bf(v);
            }
        }
}

// ---------------- fused attention:  softmax(QK^T/8) @ V ----------------
// grid (LQ/64, H, B), block 128 (4 waves); LDS score tile 64x256 (stride 257)
// V is pre-transposed per head: VT[((b*H+h)*HD+hd)*S + s]
#define SCW 257
__global__ void attn_kernel(const unsigned short* __restrict__ Q,
                            const unsigned short* __restrict__ Kb,
                            const unsigned short* __restrict__ VT,
                            unsigned short* __restrict__ CTX) {
    extern __shared__ float sc[];   // [64][SCW]
    int tid  = threadIdx.x;
    int lane = tid & 31, wv = tid >> 5;
    int hi   = lane >> 4, ln = lane & 15;
    int qbase = blockIdx.x * 64;
    int h = blockIdx.y, b = blockIdx.z;
    int r0 = wv * 16;

    // ---- phase 1: scores = Q K^T * (1/8) into LDS ----
    Frag afr[2];
    {
        const unsigned short* arow =
            Q + (size_t)(b * LQ_ + qbase + r0 + ln) * D_ + h * HD_;
#pragma unroll
        for (int kk = 0; kk < 2; ++kk)
#pragma unroll
            for (int j = 0; j < 8; ++j)
                afr[kk].u[j] = *(const unsigned int*)(arow + kk * 32 + a_kbase(j, hi));
    }
    const unsigned short* krow = Kb + (size_t)(b * S_ + ln) * D_ + h * HD_ + hi * 16;
    auto loadK = [&](Frag* d, int nt) {
#pragma unroll
        for (int kk = 0; kk < 2; ++kk)
#pragma unroll
            for (int j = 0; j < 8; ++j)
                d[kk].u[j] = *(const unsigned int*)(krow + (size_t)(nt * 16) * D_
                                                    + kk * 32 + 2 * j);
    };
    auto scoreTile = [&](Frag* kf, int nt) {
        v8f acc = {};
        acc = WMMA_BF16(afr[0].v, kf[0].v, acc);
        acc = WMMA_BF16(afr[1].v, kf[1].v, acc);
#pragma unroll
        for (int j = 0; j < 8; ++j)
            sc[(r0 + j + hi * 8) * SCW + nt * 16 + ln] = acc[j] * 0.125f;
    };
    Frag kf0[2], kf1[2];
    loadK(kf0, 0);
    for (int nt = 0; nt < 16; nt += 2) {
        loadK(kf1, nt + 1);
        scoreTile(kf0, nt);
        if (nt + 2 < 16) loadK(kf0, nt + 2);
        scoreTile(kf1, nt + 1);
    }
    __syncthreads();

    // ---- phase 2: row softmax over S=256 (one thread per row) ----
    if (tid < 64) {
        float* row = sc + tid * SCW;
        float m = row[0];
        for (int i = 1; i < 256; ++i) m = fmaxf(m, row[i]);
        float s = 0.0f;
        for (int i = 0; i < 256; ++i) { float e = __expf(row[i] - m); row[i] = e; s += e; }
        float inv = 1.0f / s;
        for (int i = 0; i < 256; ++i) row[i] *= inv;
    }
    __syncthreads();

    // ---- phase 3: ctx = P @ V  (16x64 per wave), V^T gives paired-k loads ----
    const unsigned short* vbase =
        VT + ((size_t)(b * H_ + h) * HD_ + ln) * S_ + hi * 16;
    auto loadP = [&](Frag& af, int ks) {
#pragma unroll
        for (int j = 0; j < 8; ++j) {
            int kb = a_kbase(j, hi);
            float x0 = sc[(r0 + ln) * SCW + ks + kb];
            float x1 = sc[(r0 + ln) * SCW + ks + kb + 1];
            af.u[j] = (unsigned int)f2bf(x0) | ((unsigned int)f2bf(x1) << 16);
        }
    };
    auto loadV = [&](Frag* d, int ks) {
#pragma unroll
        for (int nf = 0; nf < 4; ++nf)
#pragma unroll
            for (int j = 0; j < 8; ++j)
                d[nf].u[j] = *(const unsigned int*)(vbase + (size_t)(nf * 16) * S_
                                                    + ks + 2 * j);
    };
    v8f acc[4] = {};
    Frag pf0, pf1, vf0[4], vf1[4];
    loadP(pf0, 0);
    loadV(vf0, 0);
    for (int ks = 0; ks < S_; ks += 64) {
        loadP(pf1, ks + 32);
        loadV(vf1, ks + 32);
#pragma unroll
        for (int nf = 0; nf < 4; ++nf)
            acc[nf] = WMMA_BF16(pf0.v, vf0[nf].v, acc[nf]);
        if (ks + 64 < S_) {
            loadP(pf0, ks + 64);
            loadV(vf0, ks + 64);
        }
#pragma unroll
        for (int nf = 0; nf < 4; ++nf)
            acc[nf] = WMMA_BF16(pf1.v, vf1[nf].v, acc[nf]);
    }
#pragma unroll
    for (int nf = 0; nf < 4; ++nf)
#pragma unroll
        for (int j = 0; j < 8; ++j) {
            size_t o = (size_t)(b * LQ_ + qbase + r0 + j + hi * 8) * D_
                     + h * HD_ + nf * 16 + ln;
            CTX[o] = f2bf(acc[nf][j]);
        }
}

// ---------------- LayerNorm x2 + box proposal epilogue ----------------
__global__ void ln_box_kernel(const float* __restrict__ F,
                              const unsigned short* __restrict__ AO,
                              const float* __restrict__ g1, const float* __restrict__ be1,
                              const float* __restrict__ g2, const float* __restrict__ be2,
                              float* __restrict__ out) {
    const size_t REF = 0;
    const size_t TGT = (size_t)B_ * LQ_ * D_;
    const size_t BOX = 2 * TGT;
    int row = blockIdx.x;
    int tid = threadIdx.x;
    __shared__ float s1[256], s2[256];
    float a0 = 0.0f, a1 = 0.0f;
    for (int c = tid; c < D_; c += 256) {
        float x = F[(size_t)row * D_ + c];
        a0 += x; a1 += x * x;
    }
    s1[tid] = a0; s2[tid] = a1;
    __syncthreads();
    for (int off = 128; off > 0; off >>= 1) {
        if (tid < off) { s1[tid] += s1[tid + off]; s2[tid] += s2[tid + off]; }
        __syncthreads();
    }
    float mean = s1[0] * (1.0f / D_);
    float var  = s2[0] * (1.0f / D_) - mean * mean;
    float inv  = rsqrtf(var + LN_EPS);
    for (int c = tid; c < D_; c += 256) {
        float xn = (F[(size_t)row * D_ + c] - mean) * inv;
        out[REF + (size_t)row * D_ + c] = xn * g1[c] + be1[c];
        out[TGT + (size_t)row * D_ + c] = xn * g2[c] + be2[c];
    }
    if (tid == 0) {
        float s[4];
        for (int i = 0; i < 4; ++i) {
            float x = bf2f(AO[(size_t)row * D_ + i]);
            s[i] = 1.0f / (1.0f + __expf(-x));
        }
        float* bo = out + BOX + (size_t)row * 4;
        bo[0] = s[0] - s[2] * 0.5f;
        bo[1] = s[1] - s[3] * 0.5f;
        bo[2] = s[0] + s[2] * 0.5f;
        bo[3] = s[1] + s[3] * 0.5f;
    }
}

// ---------------- host side ----------------
extern "C" void kernel_launch(void* const* d_in, const int* in_sizes, int n_in,
                              void* d_out, int out_size, void* d_ws, size_t ws_size,
                              hipStream_t stream) {
    (void)in_sizes; (void)n_in; (void)out_size; (void)ws_size;
    const float* memory = (const float*)d_in[0];
    const float* gaze   = (const float*)d_in[1];
    const float* text   = (const float*)d_in[2];
    const float* w_in   = (const float*)d_in[3];
    const float* b_in   = (const float*)d_in[4];
    const float* w_o    = (const float*)d_in[5];
    const float* b_o    = (const float*)d_in[6];
    const float* w1     = (const float*)d_in[7];
    const float* b1     = (const float*)d_in[8];
    const float* w2     = (const float*)d_in[9];
    const float* b2     = (const float*)d_in[10];
    const float* g1     = (const float*)d_in[11];
    const float* be1    = (const float*)d_in[12];
    const float* g2     = (const float*)d_in[13];
    const float* be2    = (const float*)d_in[14];
    float* out = (float*)d_out;

    // workspace carve-up (256B aligned)
    char* ws = (char*)d_ws;
    size_t off = 0;
    auto carve = [&](size_t bytes) {
        char* p = ws + off;
        off = (off + bytes + 255) & ~(size_t)255;
        return p;
    };
    const size_t NQ = (size_t)B_ * LQ_;        // 8192 rows
    const size_t NK = (size_t)B_ * S_;         // 2048 rows
    unsigned short* Xc   = (unsigned short*)carve(NQ * D_ * 2);
    unsigned short* Xt   = (unsigned short*)carve(NK * D_ * 2);
    unsigned short* Wqkv = (unsigned short*)carve((size_t)3 * D_ * D_ * 2);
    unsigned short* Wo   = (unsigned short*)carve((size_t)D_ * D_ * 2);
    unsigned short* W1   = (unsigned short*)carve((size_t)D_ * D_ * 2);
    unsigned short* W2   = (unsigned short*)carve((size_t)D_ * D_ * 2);
    unsigned short* Qb   = (unsigned short*)carve(NQ * D_ * 2);
    unsigned short* Kb   = (unsigned short*)carve(NK * D_ * 2);
    unsigned short* VT   = (unsigned short*)carve(NK * D_ * 2);   // (B,H,HD,S)
    unsigned short* CTX  = (unsigned short*)carve(NQ * D_ * 2);
    unsigned short* AO   = (unsigned short*)carve(NQ * D_ * 2);
    unsigned short* Hf   = (unsigned short*)carve(NQ * D_ * 2);
    float*          Ff   = (float*)carve(NQ * D_ * 4);

    // 1) pack + convert to bf16
    pack_combined_kernel<<<(NQ * D_) / 256, 256, 0, stream>>>(memory, gaze, Xc);
    {
        int n = NK * D_;
        cvt_f32_bf16_kernel<<<(n + 255) / 256, 256, 0, stream>>>(text, Xt, n);
    }
    {
        int n = 3 * D_ * D_;
        cvt_f32_bf16_kernel<<<(n + 255) / 256, 256, 0, stream>>>(w_in, Wqkv, n);
    }
    {
        int n = D_ * D_;
        cvt_f32_bf16_kernel<<<(n + 255) / 256, 256, 0, stream>>>(w_o, Wo, n);
        cvt_f32_bf16_kernel<<<(n + 255) / 256, 256, 0, stream>>>(w1, W1, n);
        cvt_f32_bf16_kernel<<<(n + 255) / 256, 256, 0, stream>>>(w2, W2, n);
    }

    // 2) Q/K/V projections (WMMA)
    dim3 blk(256);
    dim3 gq(NQ / 128, D_ / 64);
    dim3 gk(NK / 128, D_ / 64);
    gemm_xwT_kernel<<<gq, blk, 0, stream>>>(Xc, Wqkv, b_in, Qb, nullptr, 0, 0);
    gemm_xwT_kernel<<<gk, blk, 0, stream>>>(Xt, Wqkv + (size_t)D_ * D_, b_in + D_,
                                            Kb, nullptr, 0, 0);
    gemm_xwT_kernel<<<gk, blk, 0, stream>>>(Xt, Wqkv + (size_t)2 * D_ * D_, b_in + 2 * D_,
                                            VT, nullptr, 0, 1);

    // 3) fused attention (WMMA + LDS softmax)
    dim3 ga(LQ_ / 64, H_, B_);
    attn_kernel<<<ga, dim3(128), 64 * SCW * 4, stream>>>(Qb, Kb, VT, CTX);

    // 4) output proj + FFN (WMMA)
    gemm_xwT_kernel<<<gq, blk, 0, stream>>>(CTX, Wo, b_o, AO, nullptr, 0, 0);
    gemm_xwT_kernel<<<gq, blk, 0, stream>>>(AO, W1, b1, Hf, nullptr, 1, 0);
    gemm_xwT_kernel<<<gq, blk, 0, stream>>>(Hf, W2, b2, nullptr, Ff, 0, 0);

    // 5) dual LayerNorm + box proposal
    ln_box_kernel<<<NQ, 256, 0, stream>>>(Ff, AO, g1, be1, g2, be2, out);
}
